// FeatureAttention_22737556865034
// MI455X (gfx1250) — matrix operand order repaired and checked by reading, not compiled, not run on hardware
//
#include <hip/hip_runtime.h>
#include <hip/hip_bf16.h>

// ---------------------------------------------------------------------------
// FeatureAttention for MI455X (gfx1250, wave32, WMMA)
//   DIM = 1024, BATCH = 256
//   Phase 1: Q/K/V = x @ W.T + b  via v_wmma_f32_16x16x32_f16 (f32 accum)
//   Phase 2: out[b,i] = sum_j softmax_j(q_i*k_j/32) * v_j   (rank-1 scores,
//            never materialize the 1 GB attn tensor; v_exp_f32 inner loop)
// ---------------------------------------------------------------------------

#define DIM   1024
#define BATCH 256

typedef __attribute__((ext_vector_type(16))) _Float16 v16h;
typedef __attribute__((ext_vector_type(8)))  _Float16 v8h;
typedef __attribute__((ext_vector_type(8)))  float    v8f;

// ---------------------------------------------------------------------------
// Kernel 0: f32 -> f16 conversion (vectorized x4)
// ---------------------------------------------------------------------------
__global__ __launch_bounds__(256) void cvt_f32_to_f16(const float* __restrict__ s,
                                                      _Float16* __restrict__ d,
                                                      int n4) {
    int i = (blockIdx.x * 256 + threadIdx.x);
    if (i < n4) {
        float4 v = ((const float4*)s)[i];
        v8h* dst8 = nullptr; (void)dst8;
        _Float16 h0 = (_Float16)v.x, h1 = (_Float16)v.y;
        _Float16 h2 = (_Float16)v.z, h3 = (_Float16)v.w;
        typedef __attribute__((ext_vector_type(4))) _Float16 v4h;
        v4h h = {h0, h1, h2, h3};
        ((v4h*)d)[i] = h;
    }
}

// ---------------------------------------------------------------------------
// Kernel 1: GEMM  Out[m,n] = sum_k Xh[m,k] * Wh[n,k] + bias[n]
//   M=256, N=1024, K=1024.  One wave computes one 16x16 tile.
//   Block = 256 threads (8 waves): tile n = (blockIdx.x*8 + wave), m = blockIdx.y.
//
//   A (16x32 f16) per-lane layout (contiguous k, same as B): lane L holds row M = L&15;
//     lanes 0-15  -> K {k..k+15}
//     lanes 16-31 -> K {k+16..k+31}
//   B (32x16 f16): lane L holds column N = L&15 (= row N of W, contiguous!);
//     lanes 0-15 -> K {k..k+15}, lanes 16-31 -> K {k+16..k+31}.
// ---------------------------------------------------------------------------
__global__ __launch_bounds__(256) void gemm_qkv_wmma(const _Float16* __restrict__ Xh,
                                                     const _Float16* __restrict__ Wh,
                                                     const float*    __restrict__ bias,
                                                     float*          __restrict__ Out) {
    const int lane = threadIdx.x & 31;
    const int wave = threadIdx.x >> 5;
    const int m0 = blockIdx.y * 16;
    const int n0 = (blockIdx.x * 8 + wave) * 16;

    const int row = lane & 15;   // M for A, N-column for B
    const int hi  = lane >> 4;   // which half-wave

    const _Float16* aptr = Xh + (size_t)(m0 + row) * DIM;
    const _Float16* bptr = Wh + (size_t)(n0 + row) * DIM;

    // Bias: every D element of this lane has N = n0 + (lane&15)
    const float bv = bias[n0 + row];
    v8f acc;
#pragma unroll
    for (int r = 0; r < 8; ++r) acc[r] = bv;

#pragma unroll 4
    for (int k = 0; k < DIM; k += 32) {
        // A: two 16-byte chunks per lane
        v8h a0 = *(const v8h*)(aptr + k + hi * 16);
        v8h a1 = *(const v8h*)(aptr + k + hi * 16 + 8);
        v16h a = __builtin_shufflevector(a0, a1,
                  0,1,2,3,4,5,6,7, 8,9,10,11,12,13,14,15);
        // B: one contiguous 32-byte run per lane
        v8h b0 = *(const v8h*)(bptr + k + hi * 16);
        v8h b1 = *(const v8h*)(bptr + k + hi * 16 + 8);
        v16h b = __builtin_shufflevector(b0, b1,
                  0,1,2,3,4,5,6,7, 8,9,10,11,12,13,14,15);
        // prefetch next K-slab
        __builtin_prefetch(aptr + k + 128, 0, 0);
        __builtin_prefetch(bptr + k + 128, 0, 0);

        acc = __builtin_amdgcn_wmma_f32_16x16x32_f16(
                  /*neg_a=*/false, a, /*neg_b=*/false, b,
                  /*c_mod=*/(short)0, acc, /*reuse_a=*/false, /*reuse_b=*/false);
    }

    // D layout: VGPR r -> (M = m0 + r + (lane>=16 ? 8 : 0), N = n0 + (lane&15))
    const int dcol  = n0 + row;
    const int drow0 = m0 + hi * 8;
#pragma unroll
    for (int r = 0; r < 8; ++r)
        Out[(size_t)(drow0 + r) * DIM + dcol] = acc[r];
}

// ---------------------------------------------------------------------------
// Kernel 2: rank-1 attention.
//   One block per batch b. K[b,:],V[b,:] interleaved in LDS (8 KB).
//   Each thread owns 4 contiguous i; inner loop walks j in pairs via b128 LDS
//   reads; 8 independent exp2 chains per iteration for ILP.
// ---------------------------------------------------------------------------
__global__ __launch_bounds__(256) void feat_attn(const float* __restrict__ Q,
                                                 const float* __restrict__ K,
                                                 const float* __restrict__ V,
                                                 float*       __restrict__ Out) {
    __shared__ float4 kvbuf[DIM / 2];   // interleaved {k,v,k,v,...}
    __shared__ float wmax[8], wmin[8];

    const int b   = blockIdx.x;
    const int tid = threadIdx.x;
    const float* Kb = K + (size_t)b * DIM;
    const float* Vb = V + (size_t)b * DIM;
    float* kvs = (float*)kvbuf;

    float kmax = -3.402823466e+38f;
    float kmin =  3.402823466e+38f;
    for (int j = tid; j < DIM; j += 256) {
        float kj = Kb[j];
        float vj = Vb[j];
        kvs[2 * j]     = kj;
        kvs[2 * j + 1] = vj;
        kmax = fmaxf(kmax, kj);
        kmin = fminf(kmin, kj);
    }
    // wave32 reduction
#pragma unroll
    for (int off = 16; off > 0; off >>= 1) {
        kmax = fmaxf(kmax, __shfl_xor(kmax, off, 32));
        kmin = fminf(kmin, __shfl_xor(kmin, off, 32));
    }
    if ((tid & 31) == 0) { wmax[tid >> 5] = kmax; wmin[tid >> 5] = kmin; }
    __syncthreads();                       // also publishes kvs[]
    kmax = wmax[0]; kmin = wmin[0];
#pragma unroll
    for (int w = 1; w < 8; ++w) {
        kmax = fmaxf(kmax, wmax[w]);
        kmin = fminf(kmin, wmin[w]);
    }

    const float LOG2E = 1.4426950408889634f;
    const float scale = 0.03125f;          // 1024^-0.5

    // 4 contiguous rows i per thread
    float4 q4 = ((const float4*)(Q + (size_t)b * DIM))[tid];
    float a[4], m[4], num[4], den[4];
    a[0] = q4.x * scale * LOG2E; a[1] = q4.y * scale * LOG2E;
    a[2] = q4.z * scale * LOG2E; a[3] = q4.w * scale * LOG2E;
#pragma unroll
    for (int u = 0; u < 4; ++u) {
        m[u]   = a[u] * (a[u] >= 0.0f ? kmax : kmin);  // max_j a*k (log2 dom.)
        num[u] = 0.0f;
        den[u] = 0.0f;
    }

#pragma unroll 2
    for (int jj = 0; jj < DIM / 2; ++jj) {
        float4 t = kvbuf[jj];              // {k0, v0, k1, v1} via ds_load_b128
#pragma unroll
        for (int u = 0; u < 4; ++u) {
            float e0 = __builtin_amdgcn_exp2f(__builtin_fmaf(a[u], t.x, -m[u]));
            float e1 = __builtin_amdgcn_exp2f(__builtin_fmaf(a[u], t.z, -m[u]));
            den[u] += e0 + e1;
            num[u]  = __builtin_fmaf(e0, t.y, num[u]);
            num[u]  = __builtin_fmaf(e1, t.w, num[u]);
        }
    }

    float4 o;
    o.x = num[0] / den[0]; o.y = num[1] / den[1];
    o.z = num[2] / den[2]; o.w = num[3] / den[3];
    ((float4*)(Out + (size_t)b * DIM))[tid] = o;
}

// ---------------------------------------------------------------------------
// Launch
// ---------------------------------------------------------------------------
extern "C" void kernel_launch(void* const* d_in, const int* in_sizes, int n_in,
                              void* d_out, int out_size, void* d_ws, size_t ws_size,
                              hipStream_t stream) {
    const float* x  = (const float*)d_in[0];
    const float* Wq = (const float*)d_in[1];
    const float* bq = (const float*)d_in[2];
    const float* Wk = (const float*)d_in[3];
    const float* bk = (const float*)d_in[4];
    const float* Wv = (const float*)d_in[5];
    const float* bv = (const float*)d_in[6];
    float* out = (float*)d_out;

    // Workspace layout (needs ~9.5 MB):
    //   Xh  : BATCH*DIM f16         (512 KB)
    //   Whq, Whk, Whv : DIM*DIM f16 (2 MB each)
    //   Q, K, V : BATCH*DIM f32     (1 MB each)
    char* ws = (char*)d_ws;
    _Float16* Xh  = (_Float16*)(ws);
    _Float16* Whq = (_Float16*)(ws + (512 << 10));
    _Float16* Whk = (_Float16*)(ws + (512 << 10) + (2u << 20));
    _Float16* Whv = (_Float16*)(ws + (512 << 10) + (4u << 20));
    float* Qb = (float*)(ws + (512 << 10) + (6u << 20));
    float* Kb = (float*)(ws + (512 << 10) + (7u << 20));
    float* Vb = (float*)(ws + (512 << 10) + (8u << 20));

    // Phase 0: convert inputs to f16
    {
        int n4x = (BATCH * DIM) / 4;            // 65536
        int n4w = (DIM * DIM) / 4;              // 262144
        cvt_f32_to_f16<<<(n4x + 255) / 256, 256, 0, stream>>>(x,  Xh,  n4x);
        cvt_f32_to_f16<<<(n4w + 255) / 256, 256, 0, stream>>>(Wq, Whq, n4w);
        cvt_f32_to_f16<<<(n4w + 255) / 256, 256, 0, stream>>>(Wk, Whk, n4w);
        cvt_f32_to_f16<<<(n4w + 255) / 256, 256, 0, stream>>>(Wv, Whv, n4w);
    }

    // Phase 1: Q/K/V GEMMs (grid: 8 n-blocks x 16 m-tiles; 8 waves/block)
    {
        dim3 grid(DIM / (16 * 8), BATCH / 16);
        gemm_qkv_wmma<<<grid, 256, 0, stream>>>(Xh, Whq, bq, Qb);
        gemm_qkv_wmma<<<grid, 256, 0, stream>>>(Xh, Whk, bk, Kb);
        gemm_qkv_wmma<<<grid, 256, 0, stream>>>(Xh, Whv, bv, Vb);
    }

    // Phase 2: rank-1 softmax attention, one block per batch sample
    feat_attn<<<BATCH, 256, 0, stream>>>(Qb, Kb, Vb, out);
}
